// MultiHeadAttention_50929722196421
// MI455X (gfx1250) — matrix operand order repaired and checked by reading, compiled, not run on hardware
//
#include <hip/hip_runtime.h>
#include <hip/hip_bf16.h>

// MI455X / gfx1250, wave32. bf16 WMMA (V_WMMA_F32_16X16X32_BF16) everywhere;
// TDM (tensor_load_to_lds + s_wait_tensorcnt) for Q-tile staging.

typedef __attribute__((ext_vector_type(16))) __bf16 v16bf;
typedef __attribute__((ext_vector_type(8)))  __bf16 v8bf;
typedef __attribute__((ext_vector_type(2)))  __bf16 v2bf;
typedef __attribute__((ext_vector_type(8)))  float  v8f;
typedef __attribute__((ext_vector_type(4)))  unsigned int v4u;
typedef __attribute__((ext_vector_type(8)))  int v8i;
typedef __attribute__((ext_vector_type(4)))  int v4i;

#define D_MODEL 1024
#define N_HEADS 16
#define DK      64
#define BATCH   4
#define SEQ     2048
#define M_ROWS  (BATCH * SEQ)   // 8192

__device__ __forceinline__ int lane_id() { return threadIdx.x & 31; }

__device__ __forceinline__ v16bf cat8(v8bf lo, v8bf hi) {
  return __builtin_shufflevector(lo, hi, 0, 1, 2, 3, 4, 5, 6, 7,
                                 8, 9, 10, 11, 12, 13, 14, 15);
}

// A fragment (16x32 bf16): per lane = one contiguous 16-elem run:
//   A[m][hf*16 + 0..15]   -> 2x b128 loads.
__device__ __forceinline__ v16bf load_a_frag(const __bf16* src, int lda) {
  const int lane = lane_id();
  const int m = lane & 15, hf = lane >> 4;
  const __bf16* p = src + m * lda + hf * 16;
  const v8bf lo = *(const v8bf*)p;
  const v8bf hi = *(const v8bf*)(p + 8);
  return cat8(lo, hi);
}

// =====================================================================
// GEMM:  C[M,N] = A[M,K](fp32) x W[K,N](fp32) + bias   via bf16 WMMA
//  out_mode 0: fp32 row-major [M,N]            (final projection)
//  out_mode 1: bf16 head layout [B,H,S,DK]     (Q, K)
//  out_mode 2: bf16 transposed head [B,H,DK,S] (V -> feeds P@V B-frags)
// block = 256 threads (8 waves); tile 256(M) x 64(N), K-chunk 32.
// Each wave owns TWO 16-row subtiles so every B-frag feeds 2 WMMAs.
// =====================================================================
#define GT_M 256
#define GT_N 64
#define GT_K 32
#define GLD_A 40   // padded LDS stride for A tile (bf16 elems, mult. of 8)

__global__ void __launch_bounds__(256)
gemm_bias_bf16wmma(const float* __restrict__ A, const float* __restrict__ W,
                   const float* __restrict__ bias,
                   float* __restrict__ outF, __bf16* __restrict__ outB,
                   int M, int N, int K, int out_mode) {
  __shared__ __bf16 sA[GT_M * GLD_A];   // row-major (A-frags contiguous)
  __shared__ __bf16 sB[4 * 512];        // fragment-major: [t][lane][j]

  const int m0   = blockIdx.x * GT_M;
  const int n0   = blockIdx.y * GT_N;
  const int tid  = threadIdx.x;
  const int wave = tid >> 5;
  const int lane = lane_id();

  v8f acc[2][4];
#pragma unroll
  for (int g = 0; g < 2; ++g)
#pragma unroll
    for (int t = 0; t < 4; ++t) acc[g][t] = v8f{};

  for (int k0 = 0; k0 < K; k0 += GT_K) {
    __syncthreads();
    // ---- stage A tile 256x32: 8 fp32 -> packed v8bf, 1x ds_store_b128
#pragma unroll
    for (int it = 0; it < 4; ++it) {
      const int row = it * 64 + (tid >> 2);
      const int col = (tid & 3) * 8;
      const float4 f0 = *(const float4*)&A[(size_t)(m0 + row) * K + k0 + col];
      const float4 f1 = *(const float4*)&A[(size_t)(m0 + row) * K + k0 + col + 4];
      v8bf pk;
      pk[0] = (__bf16)f0.x; pk[1] = (__bf16)f0.y;
      pk[2] = (__bf16)f0.z; pk[3] = (__bf16)f0.w;
      pk[4] = (__bf16)f1.x; pk[5] = (__bf16)f1.y;
      pk[6] = (__bf16)f1.z; pk[7] = (__bf16)f1.w;
      *(v8bf*)&sA[row * GLD_A + col] = pk;
    }
    // ---- stage W tile 32x64 in fragment-major order:
    // sB[t*512 + l*16 + j] = W[k0 + (l>>4)*16 + j][n0 + t*16 + (l&15)]
    {
      const int t  = tid >> 6;
      const int l  = (tid >> 1) & 31;
      const int jh = tid & 1;
      const int kr0 = (l >> 4) * 16 + jh * 8;
      const int c   = t * 16 + (l & 15);
      v8bf pk;
#pragma unroll
      for (int i = 0; i < 8; ++i)
        pk[i] = (__bf16)W[(size_t)(k0 + kr0 + i) * N + n0 + c];
      *(v8bf*)&sB[t * 512 + l * 16 + jh * 8] = pk;
    }
    if (k0 + GT_K < K)   // global_prefetch_b8 for the next A chunk
      __builtin_prefetch(&A[(size_t)(m0 + tid) * K + k0 + GT_K], 0, 0);
    __syncthreads();

    const v16bf a0 = load_a_frag(&sA[(wave * 32) * GLD_A], GLD_A);
    const v16bf a1 = load_a_frag(&sA[(wave * 32 + 16) * GLD_A], GLD_A);
#pragma unroll
    for (int t = 0; t < 4; ++t) {
      const v16bf b = *(const v16bf*)&sB[t * 512 + lane * 16];  // 2x ds_load_b128
      acc[0][t] = __builtin_amdgcn_wmma_f32_16x16x32_bf16(
          false, a0, false, b, (short)0, acc[0][t], false, false);
      acc[1][t] = __builtin_amdgcn_wmma_f32_16x16x32_bf16(
          false, a1, false, b, (short)0, acc[1][t], false, false);
    }
  }

  // epilogue: D rows M = r + 8*(lane>=16), col N = lane&15
  const int nloc = lane & 15, hf = lane >> 4;
#pragma unroll
  for (int g = 0; g < 2; ++g) {
#pragma unroll
    for (int t = 0; t < 4; ++t) {
#pragma unroll
      for (int r = 0; r < 8; ++r) {
        const int row = m0 + wave * 32 + g * 16 + r + 8 * hf;
        const int col = n0 + t * 16 + nloc;
        const float v = acc[g][t][r] + bias[col];
        if (out_mode == 0) {
          outF[(size_t)row * N + col] = v;
        } else {
          const int bb = row / SEQ, s = row % SEQ;
          const int hh = col >> 6, d = col & 63;
          if (out_mode == 1)      // [B,H,S,DK]
            outB[((size_t)(bb * N_HEADS + hh) * SEQ + s) * DK + d] = (__bf16)v;
          else                    // [B,H,DK,S]  (V transposed)
            outB[((size_t)(bb * N_HEADS + hh) * DK + d) * SEQ + s] = (__bf16)v;
        }
      }
    }
  }
}

// =====================================================================
// Attention: block = one (b,h) x one 16-row query tile. 128 thr (4 waves).
// LDS: fp32 scores 16x2048 (128KB) + bf16 probs 16x2048 (64KB) + Q tile
// (needs CDNA5's 320KB WGP LDS). Q tile is staged via the Tensor Data
// Mover (tensor_load_to_lds, TENSORcnt). QK^T B-frags load straight from
// global (contiguous in K head layout); P@V B-frags contiguous in V^T.
// =====================================================================
#define QT 16

__global__ void __launch_bounds__(128)
attention_softmax_av(const __bf16* __restrict__ Qb, const __bf16* __restrict__ Kb,
                     const __bf16* __restrict__ Vt,
                     float* __restrict__ attnOut, float* __restrict__ Octx) {
  extern __shared__ char smem[];
  float*  sc  = (float*)smem;                 // [QT][SEQ] fp32 scores
  __bf16* pb  = (__bf16*)(sc + QT * SEQ);     // [QT][SEQ] bf16 probs
  __bf16* qs  = pb + QT * SEQ;                // [QT][DK]  Q tile
  float*  red = (float*)(qs + QT * DK);       // [QT][8]
  float*  rv  = red + QT * 8;                 // [QT]

  const int nqt  = SEQ / QT;                  // 128
  const int bh   = blockIdx.x / nqt;
  const int qt   = blockIdx.x % nqt;
  const int b    = bh / N_HEADS;
  const int hh   = bh % N_HEADS;
  const int q0   = qt * QT;
  const int tid  = threadIdx.x;
  const int wave = tid >> 5;
  const int lane = tid & 31;
  const int nloc = lane & 15, hf = lane >> 4;

  const __bf16* Qh  = Qb + (size_t)bh * SEQ * DK;
  const __bf16* Kh  = Kb + (size_t)bh * SEQ * DK;
  const __bf16* Vth = Vt + (size_t)bh * DK * SEQ;

  // ---- stage Q tile (16x64 bf16 = one contiguous 1024-elem line) via TDM.
  // D# per ISA 8.3/8.4: group0 = {count=1, lds_addr, global_addr, type=2},
  // group1 = {data_size=2B, tensor_dim0=1024, tile_dim0=1024, tile_dim1=1}.
  if (wave == 0) {
    const unsigned long long ga =
        (unsigned long long)(uintptr_t)(Qh + (size_t)q0 * DK);
    const unsigned int lds_off = (unsigned int)((char*)qs - (char*)smem);
    v4u g0;
    g0[0] = 1u;                                            // count=1, user D#
    g0[1] = lds_off;                                       // lds_addr
    g0[2] = (unsigned int)(ga & 0xffffffffu);              // global_addr lo
    g0[3] = (unsigned int)((ga >> 32) & 0x01ffffffu) | (2u << 30);  // hi|type=2
    v8i g1;
    g1[0] = (1 << 16);                 // workgroup_mask=0, data_size=1 (2B)
    g1[1] = (int)(1024u << 16);        // tensor_dim0 = 1024 (bits 79:48)
    g1[2] = (int)(1u << 16);           // tensor_dim1 = 1    (bits 111:80)
    g1[3] = (int)(1024u << 16);        // tile_dim0   = 1024 (bits 127:112)
    g1[4] = 1;                         // tile_dim1 = 1, tile_dim2 = 0
    g1[5] = 1024;                      // tensor_dim0_stride (bits 207:160)
    g1[6] = (int)(1024u << 16);        // tensor_dim1_stride (bits 255:208)
    g1[7] = 0;
    const v4i gz = (v4i)0;
#if __clang_major__ >= 23
    __builtin_amdgcn_tensor_load_to_lds(g0, g1, gz, gz, (v8i)0, 0);
#else
    __builtin_amdgcn_tensor_load_to_lds(g0, g1, gz, gz, 0);
#endif
    __builtin_amdgcn_s_wait_tensorcnt(0);
  }
  __syncthreads();

  // Q A-fragments (d 0..31, 32..63), reused for all key tiles
  const v16bf a0 = load_a_frag(qs, DK);
  const v16bf a1 = load_a_frag(qs + 32, DK);

  // ---- scores = Q K^T * 1/sqrt(dk); wave covers key tiles ct = wave + 4i.
  for (int ct = wave; ct < SEQ / 16; ct += 4) {
    const int kbase = ct * 16;
    const __bf16* kp = Kh + (size_t)(kbase + nloc) * DK + hf * 16;
    const v16bf b0 = *(const v16bf*)kp;          // d = 0..31 chunk
    const v16bf b1 = *(const v16bf*)(kp + 32);   // d = 32..63 chunk
    v8f c = {};
    c = __builtin_amdgcn_wmma_f32_16x16x32_bf16(false, a0, false, b0, (short)0, c, false, false);
    c = __builtin_amdgcn_wmma_f32_16x16x32_bf16(false, a1, false, b1, (short)0, c, false, false);
#pragma unroll
    for (int r = 0; r < 8; ++r)
      sc[(r + 8 * hf) * SEQ + kbase + nloc] = c[r] * 0.125f;
  }
  __syncthreads();

  // ---- softmax (8 threads per row, float2 granularity)
  const int row = tid >> 3;
  const int sub = tid & 7;
  float* srow = sc + row * SEQ;
  float mx = -3.0e38f;
  for (int j = sub * 2; j < SEQ; j += 16) {
    const float2 v = *(const float2*)&srow[j];
    mx = fmaxf(mx, fmaxf(v.x, v.y));
  }
  red[row * 8 + sub] = mx;
  __syncthreads();
  if (sub == 0) {
    float m2 = red[row * 8];
#pragma unroll
    for (int j = 1; j < 8; ++j) m2 = fmaxf(m2, red[row * 8 + j]);
    rv[row] = m2;
  }
  __syncthreads();
  const float rmax = rv[row];
  float sum = 0.f;
  for (int j = sub * 2; j < SEQ; j += 16) {
    const float2 v = *(const float2*)&srow[j];
    const float e0 = __expf(v.x - rmax), e1 = __expf(v.y - rmax);
    *(float2*)&srow[j] = make_float2(e0, e1);
    sum += e0 + e1;
  }
  red[row * 8 + sub] = sum;
  __syncthreads();
  if (sub == 0) {
    float s2 = 0.f;
#pragma unroll
    for (int j = 0; j < 8; ++j) s2 += red[row * 8 + j];
    rv[row] = 1.0f / s2;
  }
  __syncthreads();
  const float rinv = rv[row];
  float* attnRow = attnOut + ((size_t)bh * SEQ + q0 + row) * SEQ;
  __bf16* prow = pb + row * SEQ;
  for (int j = sub * 2; j < SEQ; j += 16) {
    const float2 v = *(const float2*)&srow[j];
    const float p0 = v.x * rinv, p1 = v.y * rinv;
    *(float2*)&attnRow[j] = make_float2(p0, p1);   // materialize attn
    v2bf pp; pp[0] = (__bf16)p0; pp[1] = (__bf16)p1;
    *(v2bf*)&prow[j] = pp;                          // bf16 probs for P@V
  }
  __syncthreads();

  // ---- O_tile = P(16xS) @ V(SxDK): wave w owns cols [w*16, w*16+16).
  const int nb = wave * 16;
  v8f oc = {};
#pragma unroll 2
  for (int kb = 0; kb < SEQ; kb += 32) {
    const v16bf ap = load_a_frag(pb + kb, SEQ);                        // 2x ds_b128
    const v16bf bf = *(const v16bf*)(Vth + (size_t)(nb + nloc) * SEQ   // 2x gl_b128
                                     + kb + hf * 16);
    oc = __builtin_amdgcn_wmma_f32_16x16x32_bf16(false, ap, false, bf,
                                                 (short)0, oc, false, false);
  }
#pragma unroll
  for (int r = 0; r < 8; ++r) {
    const int m = r + 8 * hf;
    Octx[((size_t)(b * SEQ) + q0 + m) * D_MODEL + hh * DK + nb + nloc] = oc[r];
  }
}

// =====================================================================
extern "C" void kernel_launch(void* const* d_in, const int* in_sizes, int n_in,
                              void* d_out, int out_size, void* d_ws, size_t ws_size,
                              hipStream_t stream) {
  const float* x  = (const float*)d_in[0];
  const float* Wq = (const float*)d_in[1];
  const float* bq = (const float*)d_in[2];
  const float* Wk = (const float*)d_in[3];
  const float* bk = (const float*)d_in[4];
  const float* Wv = (const float*)d_in[5];
  const float* bv = (const float*)d_in[6];
  const float* Wo = (const float*)d_in[7];
  const float* bo = (const float*)d_in[8];

  float* out_p  = (float*)d_out;                            // [B,S,D]
  float* attn_p = out_p + (size_t)BATCH * SEQ * D_MODEL;    // [B,H,S,S]

  const size_t e = (size_t)BATCH * N_HEADS * SEQ * DK;
  __bf16* Qb = (__bf16*)d_ws;       // [B,H,S,DK]
  __bf16* Kb = Qb + e;              // [B,H,S,DK]
  __bf16* Vt = Kb + e;              // [B,H,DK,S]
  float*  O  = (float*)(Vt + e);    // [B,S,D]

  const dim3 ggrid(M_ROWS / GT_M, D_MODEL / GT_N);          // 32 x 16
  gemm_bias_bf16wmma<<<ggrid, 256, 0, stream>>>(x, Wq, bq, nullptr, Qb,
                                                M_ROWS, D_MODEL, D_MODEL, 1);
  gemm_bias_bf16wmma<<<ggrid, 256, 0, stream>>>(x, Wk, bk, nullptr, Kb,
                                                M_ROWS, D_MODEL, D_MODEL, 1);
  gemm_bias_bf16wmma<<<ggrid, 256, 0, stream>>>(x, Wv, bv, nullptr, Vt,
                                                M_ROWS, D_MODEL, D_MODEL, 2);

  const int    ablocks = BATCH * N_HEADS * (SEQ / QT);      // 8192
  const size_t asmem   = (size_t)QT * SEQ * 4 + (size_t)QT * SEQ * 2
                       + QT * DK * 2 + QT * 8 * 4 + QT * 4;
  attention_softmax_av<<<ablocks, 128, asmem, stream>>>(Qb, Kb, Vt, attn_p, O);

  gemm_bias_bf16wmma<<<ggrid, 256, 0, stream>>>(O, Wo, bo, out_p, nullptr,
                                                M_ROWS, D_MODEL, D_MODEL, 0);
}
